// EssenceNetClassifier_44667659878729
// MI455X (gfx1250) — matrix operand order repaired and checked by reading, not compiled, not run on hardware
//
#include <hip/hip_runtime.h>
#include <hip/hip_bf16.h>
#include <math.h>

// ---------------------------------------------------------------------------
// EssenceNet classifier for MI455X (gfx1250, wave32, WMMA)
//
//   1) gray conversion
//   2) patchify convs + BN + SiLU  -> compact per-block feature maps F
//   3) area pools (rc)             -> F (channels 0..2 of each block)
//   4) tokenize: nearest-upsample gather into bf16 token matrix [32768, 2080]
//   5) GEMM1 (bf16 WMMA, double-buffered async LDS staging): tokens @ w1 (+b1)
//   6) LayerNorm + SiLU -> Hb bf16 [32768, 1056]
//   7) GEMM2 (bf16 WMMA): Hb @ w2 -> logits f32 [32768, 1024] (+b2)
//   8) per-token argmax + L2 norm
//   9) per-sample mode + radix-select lower-median norm + copy selected row
//
// Weights pre-transposed to [N, K] so both GEMM operands stream K-contiguous:
// every WMMA fragment is two ds_load_b128's; tiles are staged by the async
// global->LDS engine (ASYNCcnt) double-buffered under the WMMA compute.
// ---------------------------------------------------------------------------

typedef __attribute__((ext_vector_type(16))) __bf16 v16bf;
typedef __attribute__((ext_vector_type(8)))  float  v8f;

#define FSAMP 326655                 // floats of F per sample
#define KP1   2080                   // padded token K  (65*32)
#define NP1   1152                   // padded hidden N (9*128)
#define KP2   1056                   // padded hidden K (33*32)
#define NP2   1024                   // padded class N  (8*128)
#define MTOT  32768                  // 2 * 128 * 128 tokens

__device__ __forceinline__ unsigned short f2bf(float f) {
  union { float f; unsigned u; } v; v.f = f;
  unsigned r = v.u + 0x7FFFu + ((v.u >> 16) & 1u);   // round-to-nearest-even
  return (unsigned short)(r >> 16);
}

// ------------------------------- gray --------------------------------------
__global__ void gray_kernel(const float* __restrict__ x, float* __restrict__ gray) {
  int idx = blockIdx.x * 256 + threadIdx.x;
  if (idx >= 2 * 65536) return;
  int b = idx >> 16, p = idx & 65535;
  const float* xb = x + (size_t)b * 3 * 65536;
  gray[idx] = 0.299f * xb[p] + 0.587f * xb[65536 + p] + 0.114f * xb[2 * 65536 + p];
}

// ---------------- patchify conv + BN + SiLU (blocks 0..6) -------------------
__global__ __launch_bounds__(256)
void conv_patch_bn_silu(const float* __restrict__ gray, const float* __restrict__ w,
                        const float* __restrict__ gamma, const float* __restrict__ beta,
                        const float* __restrict__ mean, const float* __restrict__ var,
                        float* __restrict__ F, int fofs, int oc, int oh, int ks) {
  __shared__ float red[256];
  int per = oh * oh;
  int idx = blockIdx.x;
  int b = idx / (oc * per);
  int rem = idx % (oc * per);
  int o = rem / per;
  int p = rem % per;
  int py = p / oh, px = p % oh;
  const float* gptr = gray + (size_t)b * 65536 + (size_t)(py * ks) * 256 + px * ks;
  const float* wptr = w + (size_t)o * ks * ks;
  float s = 0.f;
  int kk2 = ks * ks;
  for (int t = threadIdx.x; t < kk2; t += 256) {
    int ky = t / ks, kx = t % ks;
    s += gptr[ky * 256 + kx] * wptr[t];
  }
  red[threadIdx.x] = s; __syncthreads();
  for (int off = 128; off > 0; off >>= 1) {
    if (threadIdx.x < off) red[threadIdx.x] += red[threadIdx.x + off];
    __syncthreads();
  }
  if (threadIdx.x == 0) {
    float y = (red[0] - mean[o]) * (gamma[o] * rsqrtf(var[o] + 1e-5f)) + beta[o];
    y = y / (1.f + __expf(-y));
    F[(size_t)b * FSAMP + fofs + (size_t)(3 + o) * per + p] = y;
  }
}

// --------------------- block 7: 3x3 stride-2 pad-1 -------------------------
__global__ void conv3x3_bn_silu(const float* __restrict__ gray, const float* __restrict__ w,
                                const float* __restrict__ gamma, const float* __restrict__ beta,
                                const float* __restrict__ mean, const float* __restrict__ var,
                                float* __restrict__ F, int fofs) {
  int idx = blockIdx.x * 256 + threadIdx.x;
  if (idx >= 2 * 8 * 16384) return;
  int b = idx / (8 * 16384);
  int rem = idx % (8 * 16384);
  int o = rem / 16384;
  int p = rem % 16384;
  int py = p >> 7, px = p & 127;
  float s = 0.f;
  #pragma unroll
  for (int ky = 0; ky < 3; ++ky)
    #pragma unroll
    for (int kx = 0; kx < 3; ++kx) {
      int iy = py * 2 - 1 + ky, ix = px * 2 - 1 + kx;
      if (iy >= 0 && iy < 256 && ix >= 0 && ix < 256)
        s += gray[(size_t)b * 65536 + iy * 256 + ix] * w[o * 9 + ky * 3 + kx];
    }
  float y = (s - mean[o]) * (gamma[o] * rsqrtf(var[o] + 1e-5f)) + beta[o];
  y = y / (1.f + __expf(-y));
  F[(size_t)b * FSAMP + fofs + (size_t)(3 + o) * 16384 + p] = y;
}

// --------------------------- area pools ------------------------------------
__global__ __launch_bounds__(256)
void pool_big(const float* __restrict__ x, float* __restrict__ F, int fofs, int oh) {
  __shared__ float red[256];
  int per = oh * oh;
  int idx = blockIdx.x;                       // B*3*per workgroups
  int b = idx / (3 * per);
  int rem = idx % (3 * per);
  int c = rem / per, p = rem % per;
  int py = p / oh, px = p % oh;
  int fh = 256 / oh;
  const float* xp = x + ((size_t)b * 3 + c) * 65536 + (size_t)py * fh * 256 + px * fh;
  float s = 0.f;
  int n = fh * fh;
  for (int t = threadIdx.x; t < n; t += 256) { int dy = t / fh, dx = t % fh; s += xp[dy * 256 + dx]; }
  red[threadIdx.x] = s; __syncthreads();
  for (int off = 128; off > 0; off >>= 1) {
    if (threadIdx.x < off) red[threadIdx.x] += red[threadIdx.x + off];
    __syncthreads();
  }
  if (threadIdx.x == 0)
    F[(size_t)b * FSAMP + fofs + (size_t)c * per + p] = red[0] / (float)n;
}

__global__ void pool_small(const float* __restrict__ x, float* __restrict__ F, int fofs, int oh) {
  int per = oh * oh;
  int idx = blockIdx.x * 256 + threadIdx.x;
  if (idx >= 2 * 3 * per) return;
  int b = idx / (3 * per);
  int rem = idx % (3 * per);
  int c = rem / per, p = rem % per;
  int py = p / oh, px = p % oh;
  int fh = 256 / oh;
  const float* xp = x + ((size_t)b * 3 + c) * 65536 + (size_t)py * fh * 256 + px * fh;
  float s = 0.f;
  for (int dy = 0; dy < fh; ++dy)
    for (int dx = 0; dx < fh; ++dx) s += xp[dy * 256 + dx];
  F[(size_t)b * FSAMP + fofs + (size_t)c * per + p] = s / (float)(fh * fh);
}

// --------------------------- tokenize (bf16) -------------------------------
__global__ void tokenize(const float* __restrict__ F, unsigned short* __restrict__ tok) {
  constexpr int CST[9]  = {0, 1027, 1542, 1801, 1932, 1999, 2034, 2053, 2064};
  constexpr int FOFS[8] = {0, 1027, 3087, 7231, 15615, 32767, 68607, 146431};
  size_t idx = (size_t)blockIdx.x * 256 + threadIdx.x;
  if (idx >= (size_t)2 * 16384 * KP1) return;
  int c = (int)(idx % KP1);
  int t = (int)((idx / KP1) % 16384);
  int b = (int)(idx / ((size_t)KP1 * 16384));
  unsigned short out = 0;
  if (c < 2064) {
    int k = 0;
    #pragma unroll
    for (int i = 1; i < 8; ++i) if (c >= CST[i]) k = i;
    int within = c - CST[k];
    int oh = 1 << k;                   // block k feature resolution
    int shift = 7 - k;
    int fy = (t >> 7) >> shift;
    int fx = (t & 127) >> shift;
    float v = F[(size_t)b * FSAMP + FOFS[k] + (size_t)within * oh * oh + fy * oh + fx];
    out = f2bf(v);
  }
  tok[idx] = out;
}

// -------------- pad + convert + TRANSPOSE weights to [N, K] ----------------
__global__ void pad_wT_bf16(const float* __restrict__ W, unsigned short* __restrict__ Wt,
                            int rows /*K*/, int cols /*N*/, int prows /*Kp*/, int pcols /*Np*/) {
  size_t idx = (size_t)blockIdx.x * 256 + threadIdx.x;
  if (idx >= (size_t)prows * pcols) return;
  int n = (int)(idx / prows), k = (int)(idx % prows);
  unsigned short v = 0;
  if (k < rows && n < cols) v = f2bf(W[(size_t)k * cols + n]);
  Wt[idx] = v;                        // Wt[n * Kp + k]
}

__global__ void pad_bias(const float* __restrict__ b, float* __restrict__ bp, int n, int pn) {
  int idx = blockIdx.x * 256 + threadIdx.x;
  if (idx >= pn) return;
  bp[idx] = (idx < n) ? b[idx] : 0.f;
}

// ------------------------- bf16 WMMA GEMM ----------------------------------
// C[M,N] = A[M,K] * Bt[N,K]^T + bias[N].  M%128==0, N%128==0, K%32==0.
// 256 thr = 8 waves as 4(M) x 2(N); wave tile 32x64 = 2x4 WMMA accumulators.
// Tiles double-buffered in LDS; staged by async global->LDS b128 copies that
// run underneath the WMMA work on the other buffer (one barrier per k-step).
#define BM 128
#define BN 128
#define BK 32
#define LDT_S 40      // LDS row stride (bf16), padded vs 32 to dodge bank conflicts

__device__ __forceinline__ void async_tile_b128(const unsigned short* __restrict__ g,
                                                unsigned short* l) {
  unsigned laddr = (unsigned)(size_t)l;     // LDS aperture: low 32 bits = LDS byte addr
  asm volatile("global_load_async_to_lds_b128 %0, %1, off"
               :: "v"(laddr), "v"(g) : "memory");
}

__global__ __launch_bounds__(256)
void gemm_bf16_wmma(const unsigned short* __restrict__ A,   // [M,K]
                    const unsigned short* __restrict__ Bt,  // [N,K]
                    const float* __restrict__ bias,
                    float* __restrict__ C,
                    int M, int N, int K) {
  __shared__ unsigned short As[2][BM * LDT_S];
  __shared__ unsigned short Bs[2][BN * LDT_S];
  const int tid   = threadIdx.x;
  const int lane  = tid & 31;
  const int wave  = tid >> 5;
  const int waveM = wave & 3;
  const int waveN = wave >> 2;
  const int m0 = blockIdx.y * BM;
  const int n0 = blockIdx.x * BN;

  // this thread's two (row, seg) staging slots (4 b128 chunks per 32-bf16 row)
  const int q0   = tid * 2;
  const int row0 = q0 >> 2,       seg0 = q0 & 3;
  const int row1 = (q0 + 1) >> 2, seg1 = (q0 + 1) & 3;
  const unsigned short* ga0 = A  + (size_t)(m0 + row0) * K + seg0 * 8;
  const unsigned short* ga1 = A  + (size_t)(m0 + row1) * K + seg1 * 8;
  const unsigned short* gb0 = Bt + (size_t)(n0 + row0) * K + seg0 * 8;
  const unsigned short* gb1 = Bt + (size_t)(n0 + row1) * K + seg1 * 8;
  const int lo0 = row0 * LDT_S + seg0 * 8;
  const int lo1 = row1 * LDT_S + seg1 * 8;

  v8f acc[2][4] = {};
  const int kTiles = K / BK;

  // prologue: stage tile 0 into buffer 0
  async_tile_b128(ga0, &As[0][lo0]);
  async_tile_b128(gb0, &Bs[0][lo0]);
  async_tile_b128(ga1, &As[0][lo1]);
  async_tile_b128(gb1, &Bs[0][lo1]);
  asm volatile("s_wait_asynccnt 0x0" ::: "memory");
  __syncthreads();

  for (int kb = 0; kb < kTiles; ++kb) {
    const int cur = kb & 1;
    const int k0n = (kb + 1) * BK;
    if (kb + 1 < kTiles) {
      // stage next tile into the other buffer while we compute on this one
      async_tile_b128(ga0 + k0n, &As[cur ^ 1][lo0]);
      async_tile_b128(gb0 + k0n, &Bs[cur ^ 1][lo0]);
      async_tile_b128(ga1 + k0n, &As[cur ^ 1][lo1]);
      async_tile_b128(gb1 + k0n, &Bs[cur ^ 1][lo1]);
      if (kb + 2 < kTiles) {                 // global_prefetch_b8 two tiles ahead
        __builtin_prefetch(ga0 + k0n + BK, 0, 1);
        __builtin_prefetch(gb0 + k0n + BK, 0, 1);
      }
    }

    // A fragments: 16x32 bf16; lane m = lane&15, K-base = 8*(lane>=16)
    // dwords vi: K pairs {0,2,4,6}+base then {16,18,20,22}+base -> 2x b128
    union { unsigned u[8]; v16bf v; } afrag[2];
    const int half = lane >> 4;
    #pragma unroll
    for (int sm = 0; sm < 2; ++sm) {
      int m = waveM * 32 + sm * 16 + (lane & 15);
      #pragma unroll
      for (int vi = 0; vi < 8; ++vi) {
        int kk = ((vi < 4) ? (2 * vi) : (16 + 2 * (vi - 4))) + half * 8;
        afrag[sm].u[vi] = *reinterpret_cast<const unsigned*>(&As[cur][m * LDT_S + kk]);
      }
    }
    // B fragments: lane col n = lane&15; K pairs {0,2,4,6}+base then {16,18,20,22}+base, base = 8*(lane>=16)
    union { unsigned u[8]; v16bf v; } bfrag[4];
    #pragma unroll
    for (int sn = 0; sn < 4; ++sn) {
      int n = waveN * 64 + sn * 16 + (lane & 15);
      #pragma unroll
      for (int vi = 0; vi < 8; ++vi)
        bfrag[sn].u[vi] = *reinterpret_cast<const unsigned*>(&Bs[cur][n * LDT_S + ((vi < 4) ? (2 * vi) : (16 + 2 * (vi - 4))) + half * 8]);
    }
    #pragma unroll
    for (int sm = 0; sm < 2; ++sm)
      #pragma unroll
      for (int sn = 0; sn < 4; ++sn)
        acc[sm][sn] = __builtin_amdgcn_wmma_f32_16x16x32_bf16(
            false, afrag[sm].v, false, bfrag[sn].v, (short)0, acc[sm][sn], false, false);

    if (kb + 1 < kTiles) {
      // my LDS reads of buf[cur] must finish before iteration kb+1 overwrites
      // it with async stores; my async writes to buf[cur^1] must have landed.
      asm volatile("s_wait_dscnt 0x0" ::: "memory");
      asm volatile("s_wait_asynccnt 0x0" ::: "memory");
      __syncthreads();
    }
  }

  // epilogue: C/D layout: VGPR r -> M = r + 8*(lane>=16); N = lane&15
  #pragma unroll
  for (int sm = 0; sm < 2; ++sm)
    #pragma unroll
    for (int sn = 0; sn < 4; ++sn) {
      int gn = n0 + waveN * 64 + sn * 16 + (lane & 15);
      float bv = bias[gn];
      #pragma unroll
      for (int r = 0; r < 8; ++r) {
        int gm = m0 + waveM * 32 + sm * 16 + r + ((lane >> 4) << 3);
        C[(size_t)gm * N + gn] = acc[sm][sn][r] + bv;
      }
    }
}

// ------------------------- LayerNorm + SiLU --------------------------------
__global__ __launch_bounds__(256)
void ln_silu_bf16(const float* __restrict__ H, const float* __restrict__ lng,
                  const float* __restrict__ lnb, unsigned short* __restrict__ Hb) {
  __shared__ float r1[256], r2[256];
  int row = blockIdx.x;
  const float* hp = H + (size_t)row * NP1;
  float s = 0.f, s2 = 0.f;
  for (int c = threadIdx.x; c < 1032; c += 256) { float v = hp[c]; s += v; s2 += v * v; }
  r1[threadIdx.x] = s; r2[threadIdx.x] = s2; __syncthreads();
  for (int off = 128; off > 0; off >>= 1) {
    if (threadIdx.x < off) { r1[threadIdx.x] += r1[threadIdx.x + off]; r2[threadIdx.x] += r2[threadIdx.x + off]; }
    __syncthreads();
  }
  float mu = r1[0] / 1032.f;
  float var = r2[0] / 1032.f - mu * mu;
  float rstd = rsqrtf(var + 1e-5f);
  unsigned short* op = Hb + (size_t)row * KP2;
  for (int c = threadIdx.x; c < KP2; c += 256) {
    unsigned short o = 0;
    if (c < 1032) {
      float v = (hp[c] - mu) * rstd * lng[c] + lnb[c];
      v = v / (1.f + __expf(-v));
      o = f2bf(v);
    }
    op[c] = o;
  }
}

// -------------------- per-token argmax + L2 norm ---------------------------
__global__ __launch_bounds__(128)
void argmax_norm(const float* __restrict__ logits, int* __restrict__ pred,
                 float* __restrict__ norms) {
  __shared__ float rv[128]; __shared__ int ri[128]; __shared__ float rs[128];
  int tok = blockIdx.x;
  const float* lp = logits + (size_t)tok * NP2;
  float bm = -3.4e38f; int bi = 0; float ss = 0.f;
  for (int c = threadIdx.x; c < 1000; c += 128) {
    float v = lp[c];
    ss += v * v;
    if (v > bm) { bm = v; bi = c; }
  }
  rv[threadIdx.x] = bm; ri[threadIdx.x] = bi; rs[threadIdx.x] = ss; __syncthreads();
  for (int off = 64; off > 0; off >>= 1) {
    if (threadIdx.x < off) {
      float v2 = rv[threadIdx.x + off]; int i2 = ri[threadIdx.x + off];
      if (v2 > rv[threadIdx.x] || (v2 == rv[threadIdx.x] && i2 < ri[threadIdx.x])) {
        rv[threadIdx.x] = v2; ri[threadIdx.x] = i2;
      }
      rs[threadIdx.x] += rs[threadIdx.x + off];
    }
    __syncthreads();
  }
  if (threadIdx.x == 0) { pred[tok] = ri[0]; norms[tok] = sqrtf(rs[0]); }
}

// --------------- per-sample mode + lower-median selection ------------------
__global__ __launch_bounds__(1024)
void finalize(const int* __restrict__ pred, const float* __restrict__ norms,
              const float* __restrict__ logits, float* __restrict__ out) {
  __shared__ int hist[1000];
  __shared__ int ired[1024];
  __shared__ int ired2[1024];
  __shared__ float fred[1024];
  int b = blockIdx.x, tid = threadIdx.x;
  const int* pp = pred + b * 16384;
  const float* nn = norms + b * 16384;
  for (int i = tid; i < 1000; i += 1024) hist[i] = 0;
  __syncthreads();
  for (int i = tid; i < 16384; i += 1024) atomicAdd(&hist[pp[i]], 1);
  __syncthreads();
  // mode class: max count, smallest index on ties
  int bc = -1, bi = 0;
  for (int i = tid; i < 1000; i += 1024) { int c = hist[i]; if (c > bc) { bc = c; bi = i; } }
  ired[tid] = bc; ired2[tid] = bi; __syncthreads();
  for (int off = 512; off > 0; off >>= 1) {
    if (tid < off) {
      int c2 = ired[tid + off], i2 = ired2[tid + off];
      if (c2 > ired[tid] || (c2 == ired[tid] && i2 < ired2[tid])) { ired[tid] = c2; ired2[tid] = i2; }
    }
    __syncthreads();
  }
  int mode = ired2[0], cnt = ired[0];
  __syncthreads();
  int kth = (cnt - 1) >> 1;
  // radix-select kth smallest masked norm (norms >= 0 -> bit order == value order)
  unsigned cand = 0;
  for (int bit = 30; bit >= 0; --bit) {
    unsigned test = cand | (1u << bit);
    int cl = 0;
    for (int i = tid; i < 16384; i += 1024)
      if (pp[i] == mode && __float_as_uint(nn[i]) < test) cl++;
    ired[tid] = cl; __syncthreads();
    for (int off = 512; off > 0; off >>= 1) { if (tid < off) ired[tid] += ired[tid + off]; __syncthreads(); }
    if (ired[0] <= kth) cand = test;
    __syncthreads();
  }
  float med = __uint_as_float(cand);
  // first masked argmin |norm - med|
  float bd = 3.4e38f; int bidx = 0x7fffffff;
  for (int i = tid; i < 16384; i += 1024) {
    if (pp[i] == mode) {
      float d = fabsf(nn[i] - med);
      if (d < bd || (d == bd && i < bidx)) { bd = d; bidx = i; }
    }
  }
  fred[tid] = bd; ired2[tid] = bidx; __syncthreads();
  for (int off = 512; off > 0; off >>= 1) {
    if (tid < off) {
      float d2 = fred[tid + off]; int i2 = ired2[tid + off];
      if (d2 < fred[tid] || (d2 == fred[tid] && i2 < ired2[tid])) { fred[tid] = d2; ired2[tid] = i2; }
    }
    __syncthreads();
  }
  int sel = ired2[0];
  const float* lp = logits + ((size_t)b * 16384 + sel) * NP2;
  for (int c = tid; c < 1000; c += 1024) out[b * 1000 + c] = lp[c];
}

// ===========================================================================
extern "C" void kernel_launch(void* const* d_in, const int* in_sizes, int n_in,
                              void* d_out, int out_size, void* d_ws, size_t ws_size,
                              hipStream_t stream) {
  (void)in_sizes; (void)n_in; (void)out_size; (void)ws_size;
  static const int OC[8]   = {1024, 512, 256, 128, 64, 32, 16, 8};
  static const int KS[8]   = {256, 128, 64, 32, 16, 8, 4, 3};
  static const int OH[8]   = {1, 2, 4, 8, 16, 32, 64, 128};
  static const int FOFS[8] = {0, 1027, 3087, 7231, 15615, 32767, 68607, 146431};

  const float* x   = (const float*)d_in[0];
  const float* w1  = (const float*)d_in[41];
  const float* b1  = (const float*)d_in[42];
  const float* lng = (const float*)d_in[43];
  const float* lnb = (const float*)d_in[44];
  const float* w2  = (const float*)d_in[45];
  const float* b2  = (const float*)d_in[46];

  char* p = (char*)d_ws;
  auto alloc = [&](size_t bytes) { char* r = p; p += (bytes + 255) & ~(size_t)255; return r; };
  float*          gray   = (float*)alloc((size_t)2 * 65536 * 4);
  float*          F      = (float*)alloc((size_t)2 * FSAMP * 4);
  unsigned short* w1t    = (unsigned short*)alloc((size_t)NP1 * KP1 * 2);   // [N,K]
  float*          b1p    = (float*)alloc((size_t)NP1 * 4);
  unsigned short* w2t    = (unsigned short*)alloc((size_t)NP2 * KP2 * 2);   // [N,K]
  float*          b2p    = (float*)alloc((size_t)NP2 * 4);
  unsigned short* tokens = (unsigned short*)alloc((size_t)MTOT * KP1 * 2);  // reused as logits
  float*          logits = (float*)tokens;                                  // 134MB <= 136MB
  float*          Hpre   = (float*)alloc((size_t)MTOT * NP1 * 4);
  unsigned short* Hb     = (unsigned short*)alloc((size_t)MTOT * KP2 * 2);
  int*            pred   = (int*)alloc((size_t)MTOT * 4);
  float*          norms  = (float*)alloc((size_t)MTOT * 4);

  // 1) gray
  gray_kernel<<<(2 * 65536 + 255) / 256, 256, 0, stream>>>(x, gray);

  // 2) convs (blocks 0..6 reduction-style, block 7 direct)
  for (int k = 0; k < 7; ++k) {
    const float* wk = (const float*)d_in[1 + 5 * k];
    const float* g  = (const float*)d_in[2 + 5 * k];
    const float* be = (const float*)d_in[3 + 5 * k];
    const float* mn = (const float*)d_in[4 + 5 * k];
    const float* vr = (const float*)d_in[5 + 5 * k];
    int nblk = 2 * OC[k] * OH[k] * OH[k];
    conv_patch_bn_silu<<<nblk, 256, 0, stream>>>(gray, wk, g, be, mn, vr, F, FOFS[k], OC[k], OH[k], KS[k]);
  }
  conv3x3_bn_silu<<<(2 * 8 * 16384 + 255) / 256, 256, 0, stream>>>(
      gray, (const float*)d_in[36], (const float*)d_in[37], (const float*)d_in[38],
      (const float*)d_in[39], (const float*)d_in[40], F, FOFS[7]);

  // 3) area pools
  for (int k = 0; k < 4; ++k)
    pool_big<<<2 * 3 * OH[k] * OH[k], 256, 0, stream>>>(x, F, FOFS[k], OH[k]);
  for (int k = 4; k < 8; ++k)
    pool_small<<<(2 * 3 * OH[k] * OH[k] + 255) / 256, 256, 0, stream>>>(x, F, FOFS[k], OH[k]);

  // 4) tokenize + weight prep (transposed)
  {
    size_t ntok = (size_t)2 * 16384 * KP1;
    tokenize<<<(unsigned)((ntok + 255) / 256), 256, 0, stream>>>(F, tokens);
  }
  pad_wT_bf16<<<((KP1 * NP1) + 255) / 256, 256, 0, stream>>>(w1, w1t, 2064, 1032, KP1, NP1);
  pad_bias<<<(NP1 + 255) / 256, 256, 0, stream>>>(b1, b1p, 1032, NP1);
  pad_wT_bf16<<<((KP2 * NP2) + 255) / 256, 256, 0, stream>>>(w2, w2t, 1032, 1000, KP2, NP2);
  pad_bias<<<(NP2 + 255) / 256, 256, 0, stream>>>(b2, b2p, 1000, NP2);

  // 5) GEMM1: [32768,2080] x [2080,1152] -> Hpre
  gemm_bf16_wmma<<<dim3(NP1 / BN, MTOT / BM), 256, 0, stream>>>(tokens, w1t, b1p, Hpre, MTOT, NP1, KP1);

  // 6) LayerNorm + SiLU -> Hb bf16 [32768,1056]
  ln_silu_bf16<<<MTOT, 256, 0, stream>>>(Hpre, lng, lnb, Hb);

  // 7) GEMM2: [32768,1056] x [1056,1024] -> logits (aliases token buffer)
  gemm_bf16_wmma<<<dim3(NP2 / BN, MTOT / BM), 256, 0, stream>>>(Hb, w2t, b2p, logits, MTOT, NP2, KP2);

  // 8) per-token argmax + norm
  argmax_norm<<<MTOT, 128, 0, stream>>>(logits, pred, norms);

  // 9) mode + median selection -> output [2,1000]
  finalize<<<2, 1024, 0, stream>>>(pred, norms, logits, (float*)d_out);
}
